// AudioMambaModel_7825430413775
// MI455X (gfx1250) — matrix operand order repaired and checked by reading, not compiled, not run on hardware
//
#include <hip/hip_runtime.h>
#include <hip/hip_bf16.h>
#include <math.h>

// ---------------- model constants ----------------
#define B_    2
#define NMELS 128
#define TLEN  512
#define E_    384
#define DEPTH 4
#define NCLS  10
#define DI_   768
#define NST   16
#define DTR   24
#define KC    4
#define WP_   63
#define L0_   945
#define MID_  472
#define LSEQ  946
#define PK    256      // patch K = 16*16
#define MPAD  1920     // ceil32(B*LSEQ=1892), also covers B*L0=1890
#define DBLD  64       // padded ld for dbl (56 -> 64)

typedef __attribute__((ext_vector_type(16))) _Float16 v16h;
typedef __attribute__((ext_vector_type(8)))  _Float16 h8;
typedef __attribute__((ext_vector_type(8)))  float    v8f;

// =====================================================================
// Branch-free WMMA GEMM on padded operands.
//   C[MPAD, ldc] = act( A[MPAD, lda](f32) * W[Npad, Kpad](f16)^T + bias ) + resid
// W is pre-packed f16, zero-padded in N and K, so NO masking is needed
// anywhere. One wave computes a 32x16 tile (two 16x16 WMMAs sharing B).
// act: 0 = none, 1 = softplus
// =====================================================================
__global__ __launch_bounds__(32)
void wmma_gemm_kernel(const float* __restrict__ A, int lda,
                      const _Float16* __restrict__ W, int ldw,
                      float* __restrict__ C, int ldc,
                      int Kpad,
                      const float* __restrict__ bias,
                      const float* __restrict__ resid,
                      int act)
{
    const int lane = threadIdx.x;          // 0..31
    const int hi   = lane >> 4;            // 0 or 1
    const int l16  = lane & 15;
    const int m0   = blockIdx.y * 32;      // two stacked M-tiles
    const int n0   = blockIdx.x * 16;

    const float*    A0 = A + (size_t)(m0 + l16) * lda;        // rows m0..m0+15
    const float*    A1 = A0 + (size_t)16 * lda;               // rows m0+16..m0+31
    const _Float16* Wr = W + (size_t)(n0 + l16) * ldw;        // out-feature row

    v8f acc0 = {}, acc1 = {};

    for (int kk = 0; kk < Kpad; kk += 32) {
        const int ka = kk + hi * 8;        // A-fragment k-base (ISA 7.12.2)
        const int kb = kk + hi * 8;        // B-fragment k-base

        float4 p0 = *(const float4*)(A0 + ka);
        float4 p1 = *(const float4*)(A0 + ka + 4);
        float4 p2 = *(const float4*)(A0 + ka + 16);
        float4 p3 = *(const float4*)(A0 + ka + 20);
        float4 q0 = *(const float4*)(A1 + ka);
        float4 q1 = *(const float4*)(A1 + ka + 4);
        float4 q2 = *(const float4*)(A1 + ka + 16);
        float4 q3 = *(const float4*)(A1 + ka + 20);
        h8 b0 = *(const h8*)(Wr + kb);
        h8 b1 = *(const h8*)(Wr + kb + 16);

        __builtin_prefetch(A0 + ka + 32, 0, 1);   // global_prefetch next K-slab
        __builtin_prefetch(A1 + ka + 32, 0, 1);

        v16h a0, a1, b;
        a0[0]=(_Float16)p0.x; a0[1]=(_Float16)p0.y; a0[2]=(_Float16)p0.z; a0[3]=(_Float16)p0.w;
        a0[4]=(_Float16)p1.x; a0[5]=(_Float16)p1.y; a0[6]=(_Float16)p1.z; a0[7]=(_Float16)p1.w;
        a0[8]=(_Float16)p2.x; a0[9]=(_Float16)p2.y; a0[10]=(_Float16)p2.z; a0[11]=(_Float16)p2.w;
        a0[12]=(_Float16)p3.x; a0[13]=(_Float16)p3.y; a0[14]=(_Float16)p3.z; a0[15]=(_Float16)p3.w;
        a1[0]=(_Float16)q0.x; a1[1]=(_Float16)q0.y; a1[2]=(_Float16)q0.z; a1[3]=(_Float16)q0.w;
        a1[4]=(_Float16)q1.x; a1[5]=(_Float16)q1.y; a1[6]=(_Float16)q1.z; a1[7]=(_Float16)q1.w;
        a1[8]=(_Float16)q2.x; a1[9]=(_Float16)q2.y; a1[10]=(_Float16)q2.z; a1[11]=(_Float16)q2.w;
        a1[12]=(_Float16)q3.x; a1[13]=(_Float16)q3.y; a1[14]=(_Float16)q3.z; a1[15]=(_Float16)q3.w;
#pragma unroll
        for (int e = 0; e < 8; ++e) { b[e] = b0[e]; b[e + 8] = b1[e]; }

        acc0 = __builtin_amdgcn_wmma_f32_16x16x32_f16(false, a0, false, b,
                                                      (short)0, acc0, false, false);
        acc1 = __builtin_amdgcn_wmma_f32_16x16x32_f16(false, a1, false, b,
                                                      (short)0, acc1, false, false);
    }

    // C/D layout: VGPR r -> row m0 + r + 8*hi, col n0 + lane%16 (no masks: padded)
    const int nC = n0 + l16;
    const float bv = bias ? bias[nC] : 0.0f;
#pragma unroll
    for (int r = 0; r < 8; ++r) {
        int mC0 = m0 + r + (hi ? 8 : 0);
        int mC1 = mC0 + 16;
        float v0 = acc0[r] + bv;
        float v1 = acc1[r] + bv;
        if (act == 1) {
            v0 = (v0 > 20.0f) ? v0 : log1pf(expf(v0));
            v1 = (v1 > 20.0f) ? v1 : log1pf(expf(v1));
        }
        if (resid) {
            v0 += resid[(size_t)mC0 * ldc + nC];
            v1 += resid[(size_t)mC1 * ldc + nC];
        }
        C[(size_t)mC0 * ldc + nC] = v0;
        C[(size_t)mC1 * ldc + nC] = v1;
    }
}

// =====================================================================
// Weight pack: f32 [N,K] -> f16 [Npad,Kpad], zero-padded
// =====================================================================
__global__ void pack_w_kernel(const float* __restrict__ src, _Float16* __restrict__ dst,
                              int N, int K, int Npad, int Kpad)
{
    int idx = blockIdx.x * blockDim.x + threadIdx.x;
    if (idx >= Npad * Kpad) return;
    int n = idx / Kpad, k = idx % Kpad;
    dst[idx] = (n < N && k < K) ? (_Float16)src[(size_t)n * K + k] : (_Float16)0.0f;
}

// =====================================================================
// Block LayerNorm over a 384-wide row (blockDim.x == 384)
// =====================================================================
__device__ __forceinline__
float block_ln_384(float* red, float v, const float* w, const float* b, int e)
{
    red[e] = v; if (e < 128) red[384 + e] = 0.0f;
    __syncthreads();
    for (int s = 256; s > 0; s >>= 1) { if (e < s) red[e] += red[e + s]; __syncthreads(); }
    float mean = red[0] * (1.0f / 384.0f);
    __syncthreads();
    float d = v - mean;
    red[e] = d * d; if (e < 128) red[384 + e] = 0.0f;
    __syncthreads();
    for (int s = 256; s > 0; s >>= 1) { if (e < s) red[e] += red[e + s]; __syncthreads(); }
    float var = red[0] * (1.0f / 384.0f);
    __syncthreads();
    return d * rsqrtf(var + 1e-5f) * w[e] + b[e];
}

__global__ __launch_bounds__(E_)
void ln_kernel(const float* __restrict__ x, float* __restrict__ y,
               const float* __restrict__ w, const float* __restrict__ b)
{
    __shared__ float red[512];
    int row = blockIdx.x, e = threadIdx.x;
    float v = x[(size_t)row * E_ + e];
    y[(size_t)row * E_ + e] = block_ln_384(red, v, w, b, e);
}

// =====================================================================
// im2col for the patch conv (stride 8, 16x16 patches)
// =====================================================================
__global__ void im2col_kernel(const float* __restrict__ x, float* __restrict__ Ap)
{
    int idx = blockIdx.x * blockDim.x + threadIdx.x;
    if (idx >= B_ * L0_ * PK) return;
    int k = idx % PK;
    int p = (idx / PK) % L0_;
    int b = idx / (PK * L0_);
    int kh = k >> 4, kw = k & 15;
    int hp = p / WP_, wp = p % WP_;
    Ap[idx] = x[((size_t)b * NMELS + (hp * 8 + kh)) * TLEN + (wp * 8 + kw)];
}

// Build h = concat(p[:MID], cls, p[MID:])
__global__ void assemble_kernel(const float* __restrict__ Cp,
                                const float* __restrict__ cls,
                                float* __restrict__ h)
{
    int idx = blockIdx.x * blockDim.x + threadIdx.x;
    if (idx >= B_ * LSEQ * E_) return;
    int e = idx % E_;
    int t = (idx / E_) % LSEQ;
    int b = idx / (E_ * LSEQ);
    float v;
    if (t == MID_) v = cls[e];
    else {
        int p = (t < MID_) ? t : t - 1;
        v = Cp[((size_t)(b * L0_ + p)) * E_ + e];
    }
    h[idx] = v;
}

// =====================================================================
// Causal depthwise conv (K=4) + SiLU.  rev=1 reads the sequence reversed.
// =====================================================================
__global__ void conv_silu_kernel(const float* __restrict__ xz,
                                 const float* __restrict__ cw,
                                 const float* __restrict__ cb,
                                 float* __restrict__ xs, int rev)
{
    int idx = blockIdx.x * blockDim.x + threadIdx.x;
    if (idx >= B_ * LSEQ * DI_) return;
    int d = idx % DI_;
    int t = (idx / DI_) % LSEQ;
    int b = idx / (DI_ * LSEQ);
    float a = cb[d];
#pragma unroll
    for (int j = 0; j < KC; ++j) {
        int tl = t - (KC - 1) + j;
        if (tl < 0) continue;
        int tm = rev ? (LSEQ - 1 - tl) : tl;
        a += cw[d * KC + j] * xz[((size_t)(b * LSEQ + tm)) * (2 * DI_) + d];
    }
    xs[idx] = a / (1.0f + expf(-a));   // silu
}

// =====================================================================
// Selective scan: one thread per (b,d), 16 states in registers.
// =====================================================================
__global__ void scan_kernel(const float* __restrict__ xs,
                            const float* __restrict__ dtf,
                            const float* __restrict__ dbl,   // [.,DBLD] (dtr|B|C|pad)
                            const float* __restrict__ Alog,
                            const float* __restrict__ Dsk,
                            const float* __restrict__ xz,
                            float* __restrict__ ysum,
                            int rev, int accumulate)
{
    int idx = blockIdx.x * blockDim.x + threadIdx.x;
    if (idx >= B_ * DI_) return;
    int d = idx % DI_, b = idx / DI_;

    float Ac[NST], hst[NST];
#pragma unroll
    for (int n = 0; n < NST; ++n) { Ac[n] = -expf(Alog[d * NST + n]); hst[n] = 0.0f; }
    float Dv = Dsk[d];

    for (int t = 0; t < LSEQ; ++t) {
        size_t r = (size_t)(b * LSEQ + t);
        float dtv = dtf[r * DI_ + d];
        float u   = xs[r * DI_ + d];
        float du  = dtv * u;
        float y   = 0.0f;
#pragma unroll
        for (int n = 0; n < NST; ++n) {
            float Bn = dbl[r * DBLD + DTR + n];
            float Cn = dbl[r * DBLD + DTR + NST + n];
            hst[n] = expf(dtv * Ac[n]) * hst[n] + du * Bn;
            y += hst[n] * Cn;
        }
        y += u * Dv;
        int to = rev ? (LSEQ - 1 - t) : t;
        size_t ro = (size_t)(b * LSEQ + to);
        float z = xz[ro * (2 * DI_) + DI_ + d];
        float val = y * (z / (1.0f + expf(-z)));
        if (accumulate) ysum[ro * DI_ + d] += val;
        else            ysum[ro * DI_ + d]  = val;
    }
}

// =====================================================================
// Head: LN(normf) on MID row, LN(cls_ln), then 10-way FC.
// =====================================================================
__global__ __launch_bounds__(E_)
void head_kernel(const float* __restrict__ h,
                 const float* __restrict__ nw, const float* __restrict__ nb,
                 const float* __restrict__ cw, const float* __restrict__ cb,
                 const float* __restrict__ fw, const float* __restrict__ fb,
                 float* __restrict__ out)
{
    __shared__ float red[512];
    __shared__ float buf[E_];
    int b = blockIdx.x, e = threadIdx.x;
    float v  = h[((size_t)(b * LSEQ + MID_)) * E_ + e];
    float v1 = block_ln_384(red, v,  nw, nb, e);
    float v2 = block_ln_384(red, v1, cw, cb, e);
    buf[e] = v2;
    __syncthreads();
    if (e < NCLS) {
        float s = fb[e];
        for (int k = 0; k < E_; ++k) s += buf[k] * fw[e * E_ + k];
        out[b * NCLS + e] = s;
    }
}

// =====================================================================
// Host-side orchestration
// =====================================================================
static inline dim3 ggrid(int Npad) { return dim3(Npad / 16, MPAD / 32); }
static inline void pack_w(const float* s, _Float16* d, int N, int K, int Np, int Kp,
                          hipStream_t st)
{
    int n = Np * Kp;
    pack_w_kernel<<<(n + 255) / 256, 256, 0, st>>>(s, d, N, K, Np, Kp);
}

extern "C" void kernel_launch(void* const* d_in, const int* in_sizes, int n_in,
                              void* d_out, int out_size, void* d_ws, size_t ws_size,
                              hipStream_t stream)
{
    (void)in_sizes; (void)n_in; (void)out_size; (void)ws_size;
    const float* x        = (const float*)d_in[0];
    const float* patch_w  = (const float*)d_in[1];
    const float* patch_b  = (const float*)d_in[2];
    const float* cls_tok  = (const float*)d_in[3];
    const float* ln_w     = (const float*)d_in[4];
    const float* ln_b     = (const float*)d_in[5];
    const float* in_proj  = (const float*)d_in[6];
    const float* conv_w   = (const float*)d_in[7];
    const float* conv_b   = (const float*)d_in[8];
    const float* xproj_w  = (const float*)d_in[9];
    const float* dtproj_w = (const float*)d_in[10];
    const float* dtproj_b = (const float*)d_in[11];
    const float* A_log    = (const float*)d_in[12];
    const float* D_skip   = (const float*)d_in[13];
    const float* out_proj = (const float*)d_in[14];
    const float* normf_w  = (const float*)d_in[15];
    const float* normf_b  = (const float*)d_in[16];
    const float* cls_ln_w = (const float*)d_in[17];
    const float* cls_ln_b = (const float*)d_in[18];
    const float* fc_w     = (const float*)d_in[19];
    const float* fc_b     = (const float*)d_in[20];
    float* out = (float*)d_out;

    // ---- workspace carve-up: f32 activations (padded to MPAD rows) ----
    float* w = (float*)d_ws;
    float* Ap   = w;  w += (size_t)MPAD * PK;
    float* Cp   = w;  w += (size_t)MPAD * E_;
    float* h    = w;  w += (size_t)MPAD * E_;
    float* hn   = w;  w += (size_t)MPAD * E_;
    float* xz   = w;  w += (size_t)MPAD * 2 * DI_;
    float* xs   = w;  w += (size_t)MPAD * DI_;
    float* dbl  = w;  w += (size_t)MPAD * DBLD;
    float* dtf  = w;  w += (size_t)MPAD * DI_;
    float* ysum = w;  w += (size_t)MPAD * DI_;
    // ---- f16 packed weights ----
    _Float16* hw = (_Float16*)w;
    _Float16* Wpatch = hw;  hw += (size_t)E_ * PK;                 // 384x256
    _Float16* Win    = hw;  hw += (size_t)DEPTH * 2 * DI_ * E_;    // 4x(1536x384)
    _Float16* Wx     = hw;  hw += (size_t)DEPTH * 2 * DBLD * DI_;  // 8x(64x768)
    _Float16* Wdt    = hw;  hw += (size_t)DEPTH * 2 * DI_ * 32;    // 8x(768x32)
    _Float16* Wout   = hw;  hw += (size_t)DEPTH * E_ * DI_;        // 4x(384x768)

    // ---- pack all weights to padded f16 ----
    pack_w(patch_w,  Wpatch, E_, PK, E_, PK, stream);
    pack_w(in_proj,  Win, DEPTH * 2 * DI_, E_, DEPTH * 2 * DI_, E_, stream);
    pack_w(out_proj, Wout, DEPTH * E_, DI_, DEPTH * E_, DI_, stream);
    for (int ld = 0; ld < DEPTH * 2; ++ld) {
        pack_w(xproj_w  + (size_t)ld * 56 * DI_, Wx  + (size_t)ld * DBLD * DI_,
               56, DI_, DBLD, DI_, stream);
        pack_w(dtproj_w + (size_t)ld * DI_ * DTR, Wdt + (size_t)ld * DI_ * 32,
               DI_, DTR, DI_, 32, stream);
    }

    const int ML = B_ * LSEQ;

    // ---- patch embedding ----
    {
        int n = B_ * L0_ * PK;
        im2col_kernel<<<(n + 255) / 256, 256, 0, stream>>>(x, Ap);
        wmma_gemm_kernel<<<ggrid(E_), 32, 0, stream>>>(
            Ap, PK, Wpatch, PK, Cp, E_, PK, patch_b, nullptr, 0);
        n = B_ * LSEQ * E_;
        assemble_kernel<<<(n + 255) / 256, 256, 0, stream>>>(Cp, cls_tok, h);
    }

    // ---- Mamba layers ----
    for (int l = 0; l < DEPTH; ++l) {
        ln_kernel<<<ML, E_, 0, stream>>>(h, hn, ln_w + l * E_, ln_b + l * E_);

        wmma_gemm_kernel<<<ggrid(2 * DI_), 32, 0, stream>>>(
            hn, E_, Win + (size_t)l * 2 * DI_ * E_, E_,
            xz, 2 * DI_, E_, nullptr, nullptr, 0);

        for (int dir = 0; dir < 2; ++dir) {
            size_t ld = (size_t)(l * 2 + dir);
            int n = B_ * LSEQ * DI_;
            conv_silu_kernel<<<(n + 255) / 256, 256, 0, stream>>>(
                xz, conv_w + ld * DI_ * KC, conv_b + ld * DI_, xs, dir);

            wmma_gemm_kernel<<<ggrid(DBLD), 32, 0, stream>>>(
                xs, DI_, Wx + ld * DBLD * DI_, DI_,
                dbl, DBLD, DI_, nullptr, nullptr, 0);

            wmma_gemm_kernel<<<ggrid(DI_), 32, 0, stream>>>(
                dbl, DBLD, Wdt + ld * DI_ * 32, 32,
                dtf, DI_, 32, dtproj_b + ld * DI_, nullptr, 1);

            int nc = B_ * DI_;
            scan_kernel<<<(nc + 255) / 256, 256, 0, stream>>>(
                xs, dtf, dbl, A_log + ld * DI_ * NST, D_skip + ld * DI_,
                xz, ysum, dir, dir);
        }

        wmma_gemm_kernel<<<ggrid(E_), 32, 0, stream>>>(
            ysum, DI_, Wout + (size_t)l * E_ * DI_, DI_,
            h, E_, DI_, nullptr, h, 0);
    }

    // ---- head ----
    head_kernel<<<B_, E_, 0, stream>>>(h, normf_w, normf_b,
                                       cls_ln_w, cls_ln_b, fc_w, fc_b, out);
}
